// Model_48567490183770
// MI455X (gfx1250) — matrix operand order repaired and checked by reading, not compiled, not run on hardware
//
#include <hip/hip_runtime.h>
#include <hip/hip_bf16.h>

typedef __attribute__((ext_vector_type(16))) _Float16 v16h;
typedef __attribute__((ext_vector_type(4)))  _Float16 v4h;
typedef __attribute__((ext_vector_type(8)))  float    v8f;

#define D_MODEL 768
#define D_QKV   2304
#define D_FF    3072
#define N_LAYER 12
#define N_HEAD  12
#define HEADD   64
#define SEQ     197
#define BATCH   8
#define NTOK    (BATCH * SEQ)      // 1576
#define NPATCH  (BATCH * 196)      // 1568
#define LORA_R  8
#define AF_BN   64

// ---------------------------------------------------------------------------
// WMMA GEMM:  C[M,N] = epilogue( A[M,K] @ W[N,K]^T )
//   block tile 128x64, 8 waves, each wave 32x32 (2x2 of 16x16), K-step 32.
//   REQUIREMENTS (guaranteed by all call sites): K % 32 == 0, N % 64 == 0.
//   M may be ragged; A-loads clamp the row, stores bound-check m < M.
//   fp32 A/W are streamed with b128 loads, converted to f16 into
//   double-buffered LDS tiles; one barrier per K-step.
// epilogue modes:
//   0: (+bias)                      3: (+bias) then relu
//   1: (+bias)*s+sh                 4: (+bias)*oscale
//   2: (+bias)*s+sh + res           5: (+bias)*s+sh then quickgelu
//   6: (+bias)*s+sh + res + add2
// ---------------------------------------------------------------------------
#define BM 128
#define BN_T 64
#define BK 32
#define LDP (BK + 8)    // half-words; row pitch = 80 bytes (8B aligned)

union Frag16 { v16h h; unsigned u[8]; };

__global__ __launch_bounds__(256)
void gemm_wmma_f16(const float* __restrict__ A, const float* __restrict__ W,
                   const float* __restrict__ bias,
                   const float* __restrict__ s, const float* __restrict__ sh,
                   const float* __restrict__ res, const float* __restrict__ add2,
                   float* __restrict__ C,
                   int M, int N, int K, int ldc, int mode, float oscale)
{
    __shared__ _Float16 As[2][BM][LDP];
    __shared__ _Float16 Bs[2][BN_T][LDP];

    const int tid   = threadIdx.x;
    const int wave  = tid >> 5;
    const int lane  = tid & 31;
    const int l15   = lane & 15;
    const int lhalf = lane >> 4;

    const int m0 = blockIdx.x * BM;
    const int n0 = blockIdx.y * BN_T;

    // 8 waves: 4 along M (32 rows each) x 2 along N (32 cols each)
    const int wm = (wave & 3) * 32;
    const int wn = (wave >> 2) * 32;

    // staging coordinates (float4 granularity: 8 quads per 32-wide row)
    const int sRow = tid >> 3;        // 0..31
    const int sQc  = (tid & 7) * 4;   // 0,4,..,28

    v8f acc[2][2] = {};

    const int nk = K / BK;

    // ---- prologue: stage tile 0 into buffer 0 ----
    {
#pragma unroll
        for (int p = 0; p < 4; ++p) {
            int row = sRow + p * 32;
            int gm = m0 + row; gm = (gm < M) ? gm : (M - 1);
            float4 v = *reinterpret_cast<const float4*>(&A[(size_t)gm * K + sQc]);
            v4h h = { (_Float16)v.x, (_Float16)v.y, (_Float16)v.z, (_Float16)v.w };
            *reinterpret_cast<v4h*>(&As[0][row][sQc]) = h;
        }
#pragma unroll
        for (int p = 0; p < 2; ++p) {
            int row = sRow + p * 32;
            float4 v = *reinterpret_cast<const float4*>(&W[(size_t)(n0 + row) * K + sQc]);
            v4h h = { (_Float16)v.x, (_Float16)v.y, (_Float16)v.z, (_Float16)v.w };
            *reinterpret_cast<v4h*>(&Bs[0][row][sQc]) = h;
        }
    }
    __syncthreads();

    for (int kt = 0; kt < nk; ++kt) {
        const int cur = kt & 1;
        const int nxt = cur ^ 1;
        const bool more = (kt + 1) < nk;

        // ---- issue global loads for the next K-tile (into registers) ----
        float4 aReg[4], wReg[2];
        if (more) {
            const int kb = (kt + 1) * BK;
#pragma unroll
            for (int p = 0; p < 4; ++p) {
                int row = sRow + p * 32;
                int gm = m0 + row; gm = (gm < M) ? gm : (M - 1);
                aReg[p] = *reinterpret_cast<const float4*>(&A[(size_t)gm * K + kb + sQc]);
            }
#pragma unroll
            for (int p = 0; p < 2; ++p) {
                int row = sRow + p * 32;
                wReg[p] = *reinterpret_cast<const float4*>(&W[(size_t)(n0 + row) * K + kb + sQc]);
            }
            if (kt + 2 < nk && tid < 64)
                __builtin_prefetch(&W[(size_t)(n0 + tid) * K + (kt + 2) * BK], 0, 1);
        }

        // ---- fragments + 4 WMMAs from the current LDS buffer ----
        Frag16 af[2], bf[2];
#pragma unroll
        for (int mi = 0; mi < 2; ++mi) {
            const int arow = wm + mi * 16 + l15;
#pragma unroll
            for (int vv = 0; vv < 8; ++vv) {
                int kk = ((vv < 4) ? (2 * vv) : (16 + 2 * (vv - 4))) + 8 * lhalf;
                af[mi].u[vv] = *reinterpret_cast<const unsigned*>(&As[cur][arow][kk]);
            }
        }
#pragma unroll
        for (int ni = 0; ni < 2; ++ni) {
            const int bcol = wn + ni * 16 + l15;
#pragma unroll
            for (int vv = 0; vv < 8; ++vv) {
                int kk = ((vv < 4) ? (2 * vv) : (16 + 2 * (vv - 4))) + 8 * lhalf;
                bf[ni].u[vv] = *reinterpret_cast<const unsigned*>(&Bs[cur][bcol][kk]);
            }
        }
#pragma unroll
        for (int mi = 0; mi < 2; ++mi)
#pragma unroll
            for (int ni = 0; ni < 2; ++ni)
                acc[mi][ni] = __builtin_amdgcn_wmma_f32_16x16x32_f16(
                    false, af[mi].h, false, bf[ni].h, (short)0, acc[mi][ni], false, false);

        // ---- spill next tile regs into the other LDS buffer ----
        if (more) {
#pragma unroll
            for (int p = 0; p < 4; ++p) {
                int row = sRow + p * 32;
                v4h h = { (_Float16)aReg[p].x, (_Float16)aReg[p].y,
                          (_Float16)aReg[p].z, (_Float16)aReg[p].w };
                *reinterpret_cast<v4h*>(&As[nxt][row][sQc]) = h;
            }
#pragma unroll
            for (int p = 0; p < 2; ++p) {
                int row = sRow + p * 32;
                v4h h = { (_Float16)wReg[p].x, (_Float16)wReg[p].y,
                          (_Float16)wReg[p].z, (_Float16)wReg[p].w };
                *reinterpret_cast<v4h*>(&Bs[nxt][row][sQc]) = h;
            }
        }
        __syncthreads();
    }

    // ---- epilogue: C/D layout: lane n = l15, VGPR e -> m = e + 8*lhalf ----
    // N % 64 == 0, so n is always in range; only m needs the bound check.
#pragma unroll
    for (int mi = 0; mi < 2; ++mi) {
#pragma unroll
        for (int ni = 0; ni < 2; ++ni) {
            const int n = n0 + wn + ni * 16 + l15;
            const float bv = bias ? bias[n] : 0.0f;
            const float sv = (mode == 1 || mode == 2 || mode == 5 || mode == 6) ? s[n]  : 0.0f;
            const float hv = (mode == 1 || mode == 2 || mode == 5 || mode == 6) ? sh[n] : 0.0f;
#pragma unroll
            for (int e = 0; e < 8; ++e) {
                const int m = m0 + wm + mi * 16 + e + 8 * lhalf;
                if (m >= M) continue;
                float val = acc[mi][ni][e] + bv;
                switch (mode) {
                    case 0: break;
                    case 1: val = val * sv + hv; break;
                    case 2: val = val * sv + hv;
                            val += res[(size_t)m * ldc + n]; break;
                    case 3: val = fmaxf(val, 0.0f); break;
                    case 4: val *= oscale; break;
                    case 5: { val = val * sv + hv;
                              val = val / (1.0f + __expf(-1.702f * val)); } break;
                    case 6: val = val * sv + hv;
                            val += res[(size_t)m * ldc + n] + add2[(size_t)m * ldc + n];
                            break;
                    default: break;
                }
                C[(size_t)m * ldc + n] = val;
            }
        }
    }
}

// ---------------------------------------------------------------------------
// LayerNorm over D, one block per row (supports strided input for cls rows)
// ---------------------------------------------------------------------------
__global__ __launch_bounds__(256)
void layernorm_kernel(const float* __restrict__ in, float* __restrict__ out,
                      const float* __restrict__ g, const float* __restrict__ b,
                      int D, long in_stride, long out_stride)
{
    const int r = blockIdx.x;
    const float* x = in + (size_t)r * in_stride;
    float* y = out + (size_t)r * out_stride;

    __shared__ float red0[8], red1[8], stat[2];
    const int tid = threadIdx.x, lane = tid & 31, wave = tid >> 5;

    float sum = 0.0f, sq = 0.0f;
    for (int d = tid; d < D; d += 256) { float v = x[d]; sum += v; sq += v * v; }
    for (int off = 16; off; off >>= 1) {
        sum += __shfl_xor(sum, off, 32);
        sq  += __shfl_xor(sq,  off, 32);
    }
    if (lane == 0) { red0[wave] = sum; red1[wave] = sq; }
    __syncthreads();
    if (tid == 0) {
        float s0 = 0.0f, s1 = 0.0f;
        for (int w = 0; w < 8; ++w) { s0 += red0[w]; s1 += red1[w]; }
        float mean = s0 / D;
        float var  = s1 / D - mean * mean;
        stat[0] = mean;
        stat[1] = rsqrtf(var + 1e-5f);
    }
    __syncthreads();
    const float mean = stat[0], rstd = stat[1];
    for (int d = tid; d < D; d += 256)
        y[d] = (x[d] - mean) * rstd * g[d] + b[d];
}

// ---------------------------------------------------------------------------
// im2col for 16x16/stride-16 patch conv: P[b*196+p][c*256+iy*16+ix]
// ---------------------------------------------------------------------------
__global__ void im2col_kernel(const float* __restrict__ x, float* __restrict__ P)
{
    int idx = blockIdx.x * 256 + threadIdx.x;
    const int total = NPATCH * D_MODEL;
    if (idx >= total) return;
    int col = idx % D_MODEL;
    int row = idx / D_MODEL;
    int b = row / 196, p = row % 196;
    int py = p / 14, px = p % 14;
    int c = col / 256, rem = col % 256;
    int iy = rem / 16, ix = rem % 16;
    P[idx] = x[(((size_t)b * 3 + c) * 224 + (py * 16 + iy)) * 224 + (px * 16 + ix)];
}

// ---------------------------------------------------------------------------
// assemble tokens: cls + patches + positional embedding
// ---------------------------------------------------------------------------
__global__ void embed_kernel(const float* __restrict__ Pe, const float* __restrict__ cls,
                             const float* __restrict__ pos, float* __restrict__ h)
{
    int idx = blockIdx.x * 256 + threadIdx.x;
    const int total = NTOK * D_MODEL;
    if (idx >= total) return;
    int d = idx % D_MODEL;
    int row = idx / D_MODEL;
    int b = row / SEQ, sIdx = row % SEQ;
    float v = (sIdx == 0) ? cls[d] : Pe[((size_t)b * 196 + (sIdx - 1)) * D_MODEL + d];
    h[idx] = v + pos[(size_t)sIdx * D_MODEL + d];
}

// ---------------------------------------------------------------------------
// LoRA rank-8: dst[r, col0:col0+768] += (t_r @ la^T) @ lb^T ; one block per row
// ---------------------------------------------------------------------------
__global__ __launch_bounds__(256)
void lora_kernel(const float* __restrict__ t, const float* __restrict__ la,
                 const float* __restrict__ lb, float* __restrict__ dst,
                 int ldd, int col0)
{
    const int r = blockIdx.x;
    __shared__ float u[LORA_R];
    __shared__ float part[8][LORA_R];
    const int tid = threadIdx.x, lane = tid & 31, wave = tid >> 5;
    const float* tr = t + (size_t)r * D_MODEL;

    float p[LORA_R] = {};
    for (int d = tid; d < D_MODEL; d += 256) {
        float tv = tr[d];
#pragma unroll
        for (int j = 0; j < LORA_R; ++j) p[j] += tv * la[j * D_MODEL + d];
    }
    for (int off = 16; off; off >>= 1) {
#pragma unroll
        for (int j = 0; j < LORA_R; ++j) p[j] += __shfl_xor(p[j], off, 32);
    }
    if (lane == 0) {
#pragma unroll
        for (int j = 0; j < LORA_R; ++j) part[wave][j] = p[j];
    }
    __syncthreads();
    if (tid < LORA_R) {
        float acc = 0.0f;
        for (int w = 0; w < 8; ++w) acc += part[w][tid];
        u[tid] = acc;
    }
    __syncthreads();
    for (int d = tid; d < D_MODEL; d += 256) {
        float acc = 0.0f;
#pragma unroll
        for (int j = 0; j < LORA_R; ++j) acc += u[j] * lb[d * LORA_R + j];
        dst[(size_t)r * ldd + col0 + d] += acc;
    }
}

// ---------------------------------------------------------------------------
// attention: one wave per query row; block = 8 waves; grid (B*H, ceil(S/8))
// ---------------------------------------------------------------------------
__global__ __launch_bounds__(256)
void attn_kernel(const float* __restrict__ qkv, float* __restrict__ o)
{
    const int bh = blockIdx.x;
    const int b = bh / N_HEAD, hh = bh % N_HEAD;
    const int wave = threadIdx.x >> 5, lane = threadIdx.x & 31;
    const int q = blockIdx.y * 8 + wave;
    __shared__ float sc[8][SEQ + 3];

    if (q >= SEQ) return;
    const float scale = 0.125f;  // 1/sqrt(64)
    const float* qrow = qkv + ((size_t)(b * SEQ + q)) * D_QKV + hh * HEADD;
    const float q0 = qrow[2 * lane] * scale;
    const float q1 = qrow[2 * lane + 1] * scale;

    const float* kbase = qkv + (size_t)b * SEQ * D_QKV + D_MODEL + hh * HEADD;
    for (int k = 0; k < SEQ; ++k) {
        const float* kr = kbase + (size_t)k * D_QKV;
        float p = q0 * kr[2 * lane] + q1 * kr[2 * lane + 1];
        for (int off = 16; off; off >>= 1) p += __shfl_xor(p, off, 32);
        if (lane == 0) sc[wave][k] = p;
    }
    float mx = -1e30f;
    for (int k = lane; k < SEQ; k += 32) mx = fmaxf(mx, sc[wave][k]);
    for (int off = 16; off; off >>= 1) mx = fmaxf(mx, __shfl_xor(mx, off, 32));
    float ssum = 0.0f;
    for (int k = lane; k < SEQ; k += 32) {
        float e = __expf(sc[wave][k] - mx);
        sc[wave][k] = e;
        ssum += e;
    }
    for (int off = 16; off; off >>= 1) ssum += __shfl_xor(ssum, off, 32);
    const float inv = 1.0f / ssum;

    const float* vbase = qkv + (size_t)b * SEQ * D_QKV + 2 * D_MODEL + hh * HEADD;
    float o0 = 0.0f, o1 = 0.0f;
    for (int k = 0; k < SEQ; ++k) {
        float w = sc[wave][k];
        const float* vr = vbase + (size_t)k * D_QKV;
        o0 += w * vr[2 * lane];
        o1 += w * vr[2 * lane + 1];
    }
    float* orow = o + ((size_t)(b * SEQ + q)) * D_MODEL + hh * HEADD;
    orow[2 * lane]     = o0 * inv;
    orow[2 * lane + 1] = o1 * inv;
}

// ---------------------------------------------------------------------------
// host-side orchestration
// ---------------------------------------------------------------------------
static inline dim3 gemm_grid(int M, int N) {
    return dim3((M + BM - 1) / BM, (N + BN_T - 1) / BN_T);
}

extern "C" void kernel_launch(void* const* d_in, const int* in_sizes, int n_in,
                              void* d_out, int out_size, void* d_ws, size_t ws_size,
                              hipStream_t stream)
{
    const float* x         = (const float*)d_in[0];
    const float* conv_w    = (const float*)d_in[1];   // [768, 768] (D, 3*16*16)
    const float* cls_emb   = (const float*)d_in[2];
    const float* pos_emb   = (const float*)d_in[3];
    const float* ln_pre_g  = (const float*)d_in[4];
    const float* ln_pre_b  = (const float*)d_in[5];
    const float* ln1_g     = (const float*)d_in[6];
    const float* ln1_b     = (const float*)d_in[7];
    const float* qkv_w     = (const float*)d_in[8];
    const float* qkv_b     = (const float*)d_in[9];
    const float* ssf_ai_s  = (const float*)d_in[10];
    const float* ssf_ai_sh = (const float*)d_in[11];
    const float* lora_q_a  = (const float*)d_in[12];
    const float* lora_q_b  = (const float*)d_in[13];
    const float* lora_v_a  = (const float*)d_in[14];
    const float* lora_v_b  = (const float*)d_in[15];
    const float* out_w     = (const float*)d_in[16];
    const float* out_b     = (const float*)d_in[17];
    const float* ssf_ao_s  = (const float*)d_in[18];
    const float* ssf_ao_sh = (const float*)d_in[19];
    const float* ln2_g     = (const float*)d_in[20];
    const float* ln2_b     = (const float*)d_in[21];
    const float* mlp_in_w  = (const float*)d_in[22];
    const float* mlp_in_b  = (const float*)d_in[23];
    const float* ssf_mi_s  = (const float*)d_in[24];
    const float* ssf_mi_sh = (const float*)d_in[25];
    const float* mlp_out_w = (const float*)d_in[26];
    const float* mlp_out_b = (const float*)d_in[27];
    const float* ssf_mo_s  = (const float*)d_in[28];
    const float* ssf_mo_sh = (const float*)d_in[29];
    const float* af_ln_g   = (const float*)d_in[30];
    const float* af_ln_b   = (const float*)d_in[31];
    const float* af_down_w = (const float*)d_in[32];
    const float* af_down_b = (const float*)d_in[33];
    const float* af_up_w   = (const float*)d_in[34];
    const float* af_up_b   = (const float*)d_in[35];
    const float* ln_post_g = (const float*)d_in[36];
    const float* ln_post_b = (const float*)d_in[37];

    float* ws = (float*)d_ws;
    float* P     = ws;                                 // [1568, 768] im2col
    float* hbuf  = P     + (size_t)NPATCH * D_MODEL;   // [1576, 768] residual stream
    float* tbuf  = hbuf  + (size_t)NTOK * D_MODEL;     // [1576, 768] LN out / GEMM A
    float* qkvb  = tbuf  + (size_t)NTOK * D_MODEL;     // [1576, 2304]
    float* obuf  = qkvb  + (size_t)NTOK * D_QKV;       // [1576, 768] attn out / Pe
    float* abuf  = obuf  + (size_t)NTOK * D_MODEL;     // [1576, 768] adaptformer out
    float* mbuf  = abuf  + (size_t)NTOK * D_MODEL;     // [1576, 3072] mlp hidden
    float* adown = mbuf  + (size_t)NTOK * D_FF;        // [1576, 64]

    const dim3 blk(256);

    // ---- patch embed: im2col + GEMM(P @ conv_w^T) -> obuf ----
    im2col_kernel<<<(NPATCH * D_MODEL + 255) / 256, blk, 0, stream>>>(x, P);
    gemm_wmma_f16<<<gemm_grid(NPATCH, D_MODEL), blk, 0, stream>>>(
        P, conv_w, nullptr, nullptr, nullptr, nullptr, nullptr,
        obuf, NPATCH, D_MODEL, D_MODEL, D_MODEL, 0, 1.0f);
    embed_kernel<<<(NTOK * D_MODEL + 255) / 256, blk, 0, stream>>>(
        obuf, cls_emb, pos_emb, hbuf);
    layernorm_kernel<<<NTOK, blk, 0, stream>>>(
        hbuf, hbuf, ln_pre_g, ln_pre_b, D_MODEL, D_MODEL, D_MODEL);

    for (int i = 0; i < N_LAYER; ++i) {
        const size_t oD  = (size_t)i * D_MODEL;
        const size_t o3D = (size_t)i * D_QKV;
        const size_t oFF = (size_t)i * D_FF;

        // --- attention with SSF + LoRA(q,v) ---
        layernorm_kernel<<<NTOK, blk, 0, stream>>>(
            hbuf, tbuf, ln1_g + oD, ln1_b + oD, D_MODEL, D_MODEL, D_MODEL);
        gemm_wmma_f16<<<gemm_grid(NTOK, D_QKV), blk, 0, stream>>>(
            tbuf, qkv_w + (size_t)i * D_QKV * D_MODEL, qkv_b + o3D,
            ssf_ai_s + o3D, ssf_ai_sh + o3D, nullptr, nullptr,
            qkvb, NTOK, D_QKV, D_MODEL, D_QKV, 1, 1.0f);
        lora_kernel<<<NTOK, blk, 0, stream>>>(
            tbuf, lora_q_a + (size_t)i * LORA_R * D_MODEL,
            lora_q_b + (size_t)i * D_MODEL * LORA_R, qkvb, D_QKV, 0);
        lora_kernel<<<NTOK, blk, 0, stream>>>(
            tbuf, lora_v_a + (size_t)i * LORA_R * D_MODEL,
            lora_v_b + (size_t)i * D_MODEL * LORA_R, qkvb, D_QKV, 2 * D_MODEL);
        attn_kernel<<<dim3(BATCH * N_HEAD, (SEQ + 7) / 8), blk, 0, stream>>>(qkvb, obuf);
        gemm_wmma_f16<<<gemm_grid(NTOK, D_MODEL), blk, 0, stream>>>(
            obuf, out_w + (size_t)i * D_MODEL * D_MODEL, out_b + oD,
            ssf_ao_s + oD, ssf_ao_sh + oD, hbuf, nullptr,
            hbuf, NTOK, D_MODEL, D_MODEL, D_MODEL, 2, 1.0f);

        // --- AdaptFormer parallel branch ---
        layernorm_kernel<<<NTOK, blk, 0, stream>>>(
            hbuf, tbuf, af_ln_g + oD, af_ln_b + oD, D_MODEL, D_MODEL, D_MODEL);
        gemm_wmma_f16<<<gemm_grid(NTOK, AF_BN), blk, 0, stream>>>(
            tbuf, af_down_w + (size_t)i * AF_BN * D_MODEL, af_down_b + (size_t)i * AF_BN,
            nullptr, nullptr, nullptr, nullptr,
            adown, NTOK, AF_BN, D_MODEL, AF_BN, 3, 1.0f);
        gemm_wmma_f16<<<gemm_grid(NTOK, D_MODEL), blk, 0, stream>>>(
            adown, af_up_w + (size_t)i * D_MODEL * AF_BN, af_up_b + oD,
            nullptr, nullptr, nullptr, nullptr,
            abuf, NTOK, D_MODEL, AF_BN, D_MODEL, 4, 0.1f);

        // --- MLP with SSF (QuickGELU) ---
        layernorm_kernel<<<NTOK, blk, 0, stream>>>(
            hbuf, tbuf, ln2_g + oD, ln2_b + oD, D_MODEL, D_MODEL, D_MODEL);
        gemm_wmma_f16<<<gemm_grid(NTOK, D_FF), blk, 0, stream>>>(
            tbuf, mlp_in_w + (size_t)i * D_FF * D_MODEL, mlp_in_b + oFF,
            ssf_mi_s + oFF, ssf_mi_sh + oFF, nullptr, nullptr,
            mbuf, NTOK, D_FF, D_MODEL, D_FF, 5, 1.0f);
        gemm_wmma_f16<<<gemm_grid(NTOK, D_MODEL), blk, 0, stream>>>(
            mbuf, mlp_out_w + (size_t)i * D_MODEL * D_FF, mlp_out_b + oD,
            ssf_mo_s + oD, ssf_mo_sh + oD, hbuf, abuf,
            hbuf, NTOK, D_MODEL, D_FF, D_MODEL, 6, 1.0f);
    }

    // ---- final LN on the cls token of each batch row ----
    layernorm_kernel<<<BATCH, blk, 0, stream>>>(
        hbuf, (float*)d_out, ln_post_g, ln_post_b,
        D_MODEL, (long)SEQ * D_MODEL, D_MODEL);
}
